// QuantSqueezeExcitation_38628935860812
// MI455X (gfx1250) — 8-bit matrix operand order repaired and checked by reading, compiled, not run on hardware
//
#include <hip/hip_runtime.h>
#include <hip/hip_bf16.h>
#include <math.h>

typedef int   v8i __attribute__((ext_vector_type(8)));
typedef float v4f __attribute__((ext_vector_type(4)));

#define B_   128
#define C_   96
#define S_   24
#define HW   3136
#define HW4  784

#define WS_POOLED 16
#define WS_S      16384

__global__ void init_kernel(float* ws) {
    ((unsigned*)ws)[0] = 0u;
}

__global__ void pool_kernel(const float* __restrict__ x, float* __restrict__ ws) {
    __shared__ float red[256];
    const int bc = blockIdx.x;
    const v4f* xp = (const v4f*)(x + (size_t)bc * HW);
    float sum = 0.f;
    for (int i = threadIdx.x; i < HW4; i += 256) {
        v4f v = xp[i];
        sum += v.x + v.y + v.z + v.w;
    }
    red[threadIdx.x] = sum;
    __syncthreads();
    for (int st = 128; st > 0; st >>= 1) {
        if ((int)threadIdx.x < st) red[threadIdx.x] += red[threadIdx.x + st];
        __syncthreads();
    }
    if (threadIdx.x == 0) {
        float mean = red[0] * (1.f / (float)HW);
        ws[WS_POOLED + bc] = mean;
        atomicMax((unsigned*)ws, __float_as_uint(fabsf(mean)));
    }
}

__device__ __forceinline__ signed char quant1(float v, float inv_scale) {
    float q = rintf(v * inv_scale);
    q = fminf(fmaxf(q, -128.f), 127.f);
    return (signed char)(int)q;
}

__global__ void se_core(const float* __restrict__ w1, const float* __restrict__ b1,
                        const float* __restrict__ w2, const float* __restrict__ b2,
                        float* __restrict__ ws) {
    __shared__ int   Aq1[B_ * 32];
    __shared__ int   Bq1[32 * 32];
    __shared__ int   Aq2[B_ * 16];
    __shared__ int   Bq2[C_ * 16];
    __shared__ float hmat[B_ * S_];
    __shared__ unsigned amax[3];

    const int tid = threadIdx.x;
    if (tid < 3) amax[tid] = 0u;
    __syncthreads();

    unsigned m1 = 0u, m2 = 0u;
    for (int i = tid; i < S_ * C_; i += 256) {
        unsigned v = __float_as_uint(fabsf(w1[i])); if (v > m1) m1 = v;
    }
    for (int i = tid; i < C_ * S_; i += 256) {
        unsigned v = __float_as_uint(fabsf(w2[i])); if (v > m2) m2 = v;
    }
    atomicMax(&amax[0], m1);
    atomicMax(&amax[1], m2);
    __syncthreads();

    const float sc_a  = fmaxf(__uint_as_float(((const unsigned*)ws)[0]) / 127.f, 1e-8f);
    const float sc_w1 = fmaxf(__uint_as_float(amax[0]) / 127.f, 1e-8f);
    const float sc_w2 = fmaxf(__uint_as_float(amax[1]) / 127.f, 1e-8f);

    signed char* a1b = (signed char*)Aq1;
    const float* pooled = ws + WS_POOLED;
    for (int i = tid; i < B_ * 128; i += 256) {
        int m = i >> 7, k = i & 127;
        a1b[i] = (k < C_) ? quant1(pooled[m * C_ + k], 1.f / sc_a) : (signed char)0;
    }
    signed char* b1q = (signed char*)Bq1;
    for (int i = tid; i < 32 * 128; i += 256) {
        int n = i >> 7, k = i & 127;
        b1q[i] = (n < S_ && k < C_) ? quant1(w1[n * C_ + k], 1.f / sc_w1) : (signed char)0;
    }
    signed char* b2q = (signed char*)Bq2;
    for (int i = tid; i < C_ * 64; i += 256) {
        int n = i >> 6, k = i & 63;
        b2q[i] = (k < S_) ? quant1(w2[n * S_ + k], 1.f / sc_w2) : (signed char)0;
    }
    __syncthreads();

    const int wave = tid >> 5;
    const int lane = tid & 31;
    const int hh = lane >> 4, ln = lane & 15;
    const int mtile = wave;

    const float gs1 = sc_a * sc_w1;
    for (int nt = 0; nt < 2; ++nt) {
        v8i acc = {};
        for (int ks = 0; ks < 2; ++ks) {
            v8i av, bv;
#pragma unroll
            for (int v = 0; v < 8; ++v) {
                int ka = ks * 64 + (v >> 1) * 16 + hh * 8 + (v & 1) * 4;
                av[v] = Aq1[(mtile * 16 + ln) * 32 + (ka >> 2)];
                int kb = ks * 64 + (v >> 1) * 16 + hh * 8 + (v & 1) * 4;
                bv[v] = Bq1[(nt * 16 + ln) * 32 + (kb >> 2)];
            }
            acc = __builtin_amdgcn_wmma_i32_16x16x64_iu8(
                 true, av,  true, bv, acc,
                 false,  false);
        }
        int n = nt * 16 + ln;
        if (n < S_) {
            float bias = b1[n];
#pragma unroll
            for (int r = 0; r < 8; ++r) {
                int m = mtile * 16 + r + 8 * hh;
                hmat[m * S_ + n] = fmaxf((float)acc[r] * gs1 + bias, 0.f);
            }
        }
    }
    __syncthreads();

    unsigned mh = 0u;
    for (int i = tid; i < B_ * S_; i += 256) {
        unsigned v = __float_as_uint(fabsf(hmat[i])); if (v > mh) mh = v;
    }
    atomicMax(&amax[2], mh);
    __syncthreads();
    const float sc_h = fmaxf(__uint_as_float(amax[2]) / 127.f, 1e-8f);
    signed char* a2b = (signed char*)Aq2;
    for (int i = tid; i < B_ * 64; i += 256) {
        int m = i >> 6, k = i & 63;
        a2b[i] = (k < S_) ? quant1(hmat[m * S_ + k], 1.f / sc_h) : (signed char)0;
    }
    __syncthreads();

    const float gs2 = sc_h * sc_w2;
    float* sout = ws + WS_S;
    for (int nt = 0; nt < 6; ++nt) {
        v8i av, bv;
#pragma unroll
        for (int v = 0; v < 8; ++v) {
            int ka = (v >> 1) * 16 + hh * 8 + (v & 1) * 4;
            av[v] = Aq2[(mtile * 16 + ln) * 16 + (ka >> 2)];
            int kb = (v >> 1) * 16 + hh * 8 + (v & 1) * 4;
            bv[v] = Bq2[(nt * 16 + ln) * 16 + (kb >> 2)];
        }
        v8i acc = {};
        acc = __builtin_amdgcn_wmma_i32_16x16x64_iu8(true, av, true, bv, acc, false, false);
        int n = nt * 16 + ln;
        float bias = b2[n];
#pragma unroll
        for (int r = 0; r < 8; ++r) {
            int m = mtile * 16 + r + 8 * hh;
            float v2 = (float)acc[r] * gs2 + bias;
            sout[m * C_ + n] = fminf(fmaxf(v2 + 3.f, 0.f), 6.f) * (1.f / 6.f);
        }
    }
}

__global__ void excite_kernel(const float* __restrict__ x,
                              const float* __restrict__ ws,
                              float* __restrict__ out) {
    int idx = blockIdx.x * 256 + threadIdx.x;
    const int total4 = B_ * C_ * HW4;
    if (idx >= total4) return;
    int bc = idx / HW4;
    float s = ws[WS_S + bc];
    v4f v = ((const v4f*)x)[idx];
    v *= s;
    __builtin_nontemporal_store(v, &((v4f*)out)[idx]);
}

extern "C" void kernel_launch(void* const* d_in, const int* in_sizes, int n_in,
                              void* d_out, int out_size, void* d_ws, size_t ws_size,
                              hipStream_t stream) {
    const float* x  = (const float*)d_in[0];
    const float* w1 = (const float*)d_in[1];
    const float* b1 = (const float*)d_in[2];
    const float* w2 = (const float*)d_in[3];
    const float* b2 = (const float*)d_in[4];
    float* ws  = (float*)d_ws;
    float* out = (float*)d_out;

    init_kernel<<<1, 1, 0, stream>>>(ws);
    pool_kernel<<<B_ * C_, 256, 0, stream>>>(x, ws);
    se_core<<<1, 256, 0, stream>>>(w1, b1, w2, b2, ws);
    const int total4 = B_ * C_ * HW4;
    excite_kernel<<<(total4 + 255) / 256, 256, 0, stream>>>(x, ws, out);
}
